// MoELayer_76244259439331
// MI455X (gfx1250) — matrix operand order repaired and checked by reading, not compiled, not run on hardware
//
#include <hip/hip_runtime.h>
#include <stdint.h>

// ---------------------------------------------------------------------------
// MoE FFN (top-2 of 8 experts), D_MODEL=1024, D_FF=4096, T=4096 tokens.
// bf16 WMMA (v_wmma_f32_16x16x32_bf16) for both GEMMs, with LDS double
// buffering filled by gfx1250 async copies (global_load_async_to_lds_b128,
// ASYNCcnt). Weights pre-transposed+converted to bf16 so all tile copies are
// contiguous 16B chunks. Experts serialized on the stream => no atomics.
// ---------------------------------------------------------------------------

typedef __bf16 v16bf __attribute__((ext_vector_type(16)));
typedef __bf16 v8bf  __attribute__((ext_vector_type(8)));
typedef float  v8f   __attribute__((ext_vector_type(8)));

#define D_MODEL 1024
#define D_FF    4096
#define NEXP    8
#define TOK     4096   // B*S = 4*1024

// GEMM block tile
#define BM 64
#define BN 128
#define BK 64
#define LDSS 72        // LDS row stride in elements (64 + 8 pad; 144B keeps 16B
                       // chunk alignment and spreads lanes across all 64 banks)

static __device__ __forceinline__ unsigned short f32_to_bf16_rne(float f) {
    unsigned int u = __float_as_uint(f);
    unsigned int r = 0x7FFFu + ((u >> 16) & 1u);
    return (unsigned short)((u + r) >> 16);
}

// ---- gfx1250 async copy: 16 bytes global -> LDS, tracked by ASYNCcnt ------
static __device__ __forceinline__ void async_cp16(unsigned short* lds_dst,
                                                  const unsigned short* gsrc) {
    unsigned loff = (unsigned)(uintptr_t)lds_dst;            // low 32b = LDS byte offset
    unsigned long long g = (unsigned long long)(uintptr_t)gsrc;
    asm volatile("global_load_async_to_lds_b128 %0, %1, off"
                 :: "v"(loff), "v"(g) : "memory");
}
static __device__ __forceinline__ void wait_async0() {
    asm volatile("s_wait_asynccnt 0x0" ::: "memory");
}

// ------------------------------ zero output --------------------------------
__global__ void zero_f32_kernel(float* __restrict__ p, int n) {
    int i = blockIdx.x * blockDim.x + threadIdx.x;
    if (i < n) p[i] = 0.0f;
}

// ------------------------------ gating -------------------------------------
__global__ void gate_kernel(const float* __restrict__ x,
                            const float* __restrict__ gw,
                            const float* __restrict__ gb,
                            float* __restrict__ cw) {
    const int lane = threadIdx.x & 31;
    const int t = blockIdx.x * (blockDim.x >> 5) + (threadIdx.x >> 5);
    if (t >= TOK) return;

    float acc[NEXP];
#pragma unroll
    for (int e = 0; e < NEXP; ++e) acc[e] = 0.0f;

    const float* xr = x + (size_t)t * D_MODEL;
    for (int k = lane; k < D_MODEL; k += 32) {
        float xv = xr[k];
        const float* g = gw + (size_t)k * NEXP;
#pragma unroll
        for (int e = 0; e < NEXP; ++e) acc[e] = fmaf(xv, g[e], acc[e]);
    }
#pragma unroll
    for (int off = 16; off > 0; off >>= 1) {
#pragma unroll
        for (int e = 0; e < NEXP; ++e) acc[e] += __shfl_xor(acc[e], off, 32);
    }
    if (lane == 0) {
        float lg[NEXP];
#pragma unroll
        for (int e = 0; e < NEXP; ++e) lg[e] = acc[e] + gb[e];
        float best = lg[0]; int bi = 0;
#pragma unroll
        for (int e = 1; e < NEXP; ++e) if (lg[e] > best) { best = lg[e]; bi = e; }
        float second = -3.4e38f; int si = 0;
#pragma unroll
        for (int e = 0; e < NEXP; ++e)
            if (e != bi && lg[e] > second) { second = lg[e]; si = e; }
        float e1 = __expf(second - best);
        float inv = 1.0f / (1.0f + e1);
        float* row = cw + (size_t)t * NEXP;
#pragma unroll
        for (int e = 0; e < NEXP; ++e) row[e] = 0.0f;
        row[bi] = inv;
        row[si] = e1 * inv;
    }
}

// ------------------------- fp32 -> bf16 convert ----------------------------
__global__ void cvt_bf16_kernel(const float* __restrict__ src,
                                unsigned short* __restrict__ dst, int n) {
    int i = blockIdx.x * blockDim.x + threadIdx.x;
    if (i < n) dst[i] = f32_to_bf16_rne(src[i]);
}

// ---------------- LDS-tiled transpose + convert (per expert) ---------------
__global__ void transpose_cvt_kernel(const float* __restrict__ src,
                                     unsigned short* __restrict__ dst,
                                     int rows, int cols) {
    __shared__ unsigned short tile[32][33];
    const size_t esz = (size_t)rows * cols;
    const float* s = src + esz * blockIdx.z;
    unsigned short* d = dst + esz * blockIdx.z;
    const int c0 = blockIdx.x * 32;
    const int r0 = blockIdx.y * 32;
    const int tx = threadIdx.x;
    const int ty = threadIdx.y;
#pragma unroll
    for (int i = 0; i < 32; i += 8)
        tile[ty + i][tx] = f32_to_bf16_rne(s[(size_t)(r0 + ty + i) * cols + (c0 + tx)]);
    __syncthreads();
#pragma unroll
    for (int i = 0; i < 32; i += 8)
        d[(size_t)(c0 + ty + i) * rows + (r0 + tx)] = tile[tx][ty + i];
}

// --------------------- WMMA core with async LDS pipeline -------------------
// Block = 256 threads (8 waves) computes BM x BN. Each wave owns 32x32
// (2x2 f32 accumulators). K advances in BK=64 stages, double-buffered in LDS:
// stage s+1 is issued with async-to-LDS copies while stage s feeds WMMAs.
static __device__ __forceinline__ void gemm_lds_core(
    const unsigned short* __restrict__ Ag, int lda,   // [M][K] bf16
    const unsigned short* __restrict__ Bg, int ldb,   // [N][K] bf16 (transposed weights)
    int K, int mblk, int nblk,
    unsigned short* sA, unsigned short* sB,           // 2*BM*LDSS / 2*BN*LDSS
    v8f acc[2][2]) {
    const int tid  = threadIdx.x;
    const int lane = tid & 31;
    const int half = lane >> 4;
    const int ml   = lane & 15;
    const int wave = tid >> 5;
    const int wm   = (wave & 1) * 32;
    const int wn   = (wave >> 1) * 32;

    // 16B-chunk assignments for the async fills (8 chunks per 64-elem row)
    int ar[2], ac[2], br[4], bc[4];
#pragma unroll
    for (int r = 0; r < 2; ++r) { int c = tid + 256 * r; ar[r] = c >> 3; ac[r] = (c & 7) * 8; }
#pragma unroll
    for (int r = 0; r < 4; ++r) { int c = tid + 256 * r; br[r] = c >> 3; bc[r] = (c & 7) * 8; }

    const int S = K / BK;

    auto issue = [&](int s) {
        unsigned short* dA = sA + (s & 1) * (BM * LDSS);
        unsigned short* dB = sB + (s & 1) * (BN * LDSS);
        const unsigned short* gA = Ag + (size_t)mblk * lda + s * BK;
        const unsigned short* gB = Bg + (size_t)nblk * ldb + s * BK;
#pragma unroll
        for (int r = 0; r < 2; ++r)
            async_cp16(dA + ar[r] * LDSS + ac[r], gA + (size_t)ar[r] * lda + ac[r]);
#pragma unroll
        for (int r = 0; r < 4; ++r)
            async_cp16(dB + br[r] * LDSS + bc[r], gB + (size_t)br[r] * ldb + bc[r]);
    };

    issue(0);
    for (int s = 0; s < S; ++s) {
        wait_async0();          // own stage-s copies landed in LDS
        __syncthreads();        // everyone's stage-s copies visible; stage s-1
                                // compute finished in all waves -> old buf free
        if (s + 1 < S) issue(s + 1);   // overlap next fill with this compute

        const unsigned short* cA = sA + (s & 1) * (BM * LDSS);
        const unsigned short* cB = sB + (s & 1) * (BN * LDSS);
#pragma unroll
        for (int ks = 0; ks < 2; ++ks) {
            const int kl = ks * 32;
            v16bf aF[2], bF[2];
#pragma unroll
            for (int i = 0; i < 2; ++i) {
                const __bf16* p = (const __bf16*)(cA + (wm + i * 16 + ml) * LDSS + kl + half * 8);
                v8bf lo = *(const v8bf*)p;
                v8bf hi = *(const v8bf*)(p + 16);
#pragma unroll
                for (int j = 0; j < 8; ++j) { aF[i][j] = lo[j]; aF[i][j + 8] = hi[j]; }
            }
#pragma unroll
            for (int i = 0; i < 2; ++i) {
                const __bf16* p = (const __bf16*)(cB + (wn + i * 16 + ml) * LDSS + kl + half * 8);
                v8bf lo = *(const v8bf*)p;
                v8bf hi = *(const v8bf*)(p + 16);
#pragma unroll
                for (int j = 0; j < 8; ++j) { bF[i][j] = lo[j]; bF[i][j + 8] = hi[j]; }
            }
#pragma unroll
            for (int i = 0; i < 2; ++i)
#pragma unroll
                for (int j = 0; j < 2; ++j)
                    acc[i][j] = __builtin_amdgcn_wmma_f32_16x16x32_bf16(
                        false, aF[i], false, bF[j], (short)0, acc[i][j], false, false);
        }
    }
}

// ---------------- GEMM1: h = relu(xb @ w1^T_e + b1_e), bf16 out ------------
__global__ __launch_bounds__(256)
void ffn1_kernel(const unsigned short* __restrict__ xb,
                 const unsigned short* __restrict__ w1T_e,  // [D_FF][D_MODEL]
                 const float* __restrict__ b1_e,
                 unsigned short* __restrict__ h) {
    __shared__ __align__(16) unsigned short sA[2 * BM * LDSS];
    __shared__ __align__(16) unsigned short sB[2 * BN * LDSS];
    const int lane = threadIdx.x & 31;
    const int wave = threadIdx.x >> 5;

    v8f acc[2][2] = {};
    gemm_lds_core(xb, D_MODEL, w1T_e, D_MODEL, D_MODEL,
                  blockIdx.x * BM, blockIdx.y * BN, sA, sB, acc);

    const int m0 = blockIdx.x * BM + (wave & 1) * 32;
    const int n0 = blockIdx.y * BN + (wave >> 1) * 32;
    const int half = lane >> 4;
    const int nl   = lane & 15;
#pragma unroll
    for (int i = 0; i < 2; ++i)
#pragma unroll
        for (int j = 0; j < 2; ++j)
#pragma unroll
            for (int r = 0; r < 8; ++r) {
                int m = m0 + i * 16 + r + 8 * half;
                int n = n0 + j * 16 + nl;
                float v = acc[i][j][r] + b1_e[n];
                v = v > 0.0f ? v : 0.0f;
                h[(size_t)m * D_FF + n] = f32_to_bf16_rne(v);
            }
}

// ------- GEMM2: out += cw[:,e] * (h @ w2^T_e + b2_e), fp32 accumulate ------
__global__ __launch_bounds__(256)
void ffn2_kernel(const unsigned short* __restrict__ h,
                 const unsigned short* __restrict__ w2T_e,  // [D_MODEL][D_FF]
                 const float* __restrict__ b2_e,
                 const float* __restrict__ cw, int expert,
                 float* __restrict__ out) {
    __shared__ __align__(16) unsigned short sA[2 * BM * LDSS];
    __shared__ __align__(16) unsigned short sB[2 * BN * LDSS];
    const int lane = threadIdx.x & 31;
    const int wave = threadIdx.x >> 5;

    v8f acc[2][2] = {};
    gemm_lds_core(h, D_FF, w2T_e, D_FF, D_FF,
                  blockIdx.x * BM, blockIdx.y * BN, sA, sB, acc);

    const int m0 = blockIdx.x * BM + (wave & 1) * 32;
    const int n0 = blockIdx.y * BN + (wave >> 1) * 32;
    const int half = lane >> 4;
    const int nl   = lane & 15;
#pragma unroll
    for (int i = 0; i < 2; ++i)
#pragma unroll
        for (int j = 0; j < 2; ++j)
#pragma unroll
            for (int r = 0; r < 8; ++r) {
                int m = m0 + i * 16 + r + 8 * half;
                int n = n0 + j * 16 + nl;
                float g = cw[(size_t)m * NEXP + expert];
                out[(size_t)m * D_MODEL + n] += g * (acc[i][j][r] + b2_e[n]);
            }
}

// ---------------------------------------------------------------------------
extern "C" void kernel_launch(void* const* d_in, const int* in_sizes, int n_in,
                              void* d_out, int out_size, void* d_ws, size_t ws_size,
                              hipStream_t stream) {
    const float* x      = (const float*)d_in[0];  // [4096,1024]
    const float* gate_w = (const float*)d_in[1];  // [1024,8]
    const float* gate_b = (const float*)d_in[2];  // [8]
    const float* w1     = (const float*)d_in[3];  // [8,1024,4096]
    const float* b1     = (const float*)d_in[4];  // [8,4096]
    const float* w2     = (const float*)d_in[5];  // [8,4096,1024]
    const float* b2     = (const float*)d_in[6];  // [8,1024]
    float* out = (float*)d_out;

    char* ws = (char*)d_ws;
    float*          cw  = (float*)(ws + 0);                   // 128 KB
    unsigned short* xb  = (unsigned short*)(ws + 131072);     // 8 MB
    unsigned short* w1T = (unsigned short*)(ws + 8519680);    // 64 MB
    unsigned short* w2T = (unsigned short*)(ws + 75628544);   // 64 MB
    unsigned short* h   = (unsigned short*)(ws + 142737408);  // 32 MB

    {
        int n = TOK * D_MODEL;
        zero_f32_kernel<<<(n + 255) / 256, 256, 0, stream>>>(out, n);
    }
    gate_kernel<<<TOK / 8, 256, 0, stream>>>(x, gate_w, gate_b, cw);
    {
        int n = TOK * D_MODEL;
        cvt_bf16_kernel<<<(n + 255) / 256, 256, 0, stream>>>(x, xb, n);
    }
    {
        dim3 blk(32, 8);
        dim3 g1(D_FF / 32, D_MODEL / 32, NEXP);
        transpose_cvt_kernel<<<g1, blk, 0, stream>>>(w1, w1T, D_MODEL, D_FF);
        dim3 g2(D_MODEL / 32, D_FF / 32, NEXP);
        transpose_cvt_kernel<<<g2, blk, 0, stream>>>(w2, w2T, D_FF, D_MODEL);
    }
    for (int e = 0; e < NEXP; ++e) {
        const unsigned short* w1Te = w1T + (size_t)e * D_FF * D_MODEL;
        const unsigned short* w2Te = w2T + (size_t)e * D_MODEL * D_FF;
        const float* b1e = b1 + (size_t)e * D_FF;
        const float* b2e = b2 + (size_t)e * D_MODEL;

        dim3 g1(TOK / BM, D_FF / BN);      // 64 x 32
        ffn1_kernel<<<g1, 256, 0, stream>>>(xb, w1Te, b1e, h);

        dim3 g2(TOK / BM, D_MODEL / BN);   // 64 x 8
        ffn2_kernel<<<g2, 256, 0, stream>>>(h, w2Te, b2e, cw, e, out);
    }
}
